// LoRALinear_27023934227119
// MI455X (gfx1250) — matrix operand order repaired and checked by reading, not compiled, not run on hardware
//
#include <hip/hip_runtime.h>
#include <cstdint>
#include <cstddef>

// ---------------------------------------------------------------------------
// LoRA linear on MI455X (gfx1250): out = x @ W^T + bias + (x@A^T)@B^T * 1/16
// Base GEMM via V_WMMA_F32_16X16X32_BF16 with bf16 hi/lo operand split
// (3 WMMAs/tile => ~fp32 accuracy at bf16 matrix-core rate).
// Epilogue tiles staged into LDS with the Tensor Data Mover (TENSOR_LOAD_TO_LDS).
// ---------------------------------------------------------------------------

typedef __bf16 bf16;
typedef bf16     v4bf __attribute__((ext_vector_type(4)));
typedef bf16     v8bf __attribute__((ext_vector_type(8)));
typedef bf16     v16bf __attribute__((ext_vector_type(16)));
typedef float    v8f  __attribute__((ext_vector_type(8)));
typedef uint32_t v4u  __attribute__((ext_vector_type(4)));
typedef int      v4i  __attribute__((ext_vector_type(4)));
typedef int      v8i  __attribute__((ext_vector_type(8)));

#define M_DIM 8192
#define N_DIM 4096
#define K_DIM 4096
#define RANK  16
#define LORA_SCALE (1.0f / 16.0f)

#define BM 128
#define BN 128
#define BK 32
#define LDA 40            // LDS row stride in bf16 elems (80B): conflict-free b128 frag loads
#define NK (K_DIM / BK)   // 128 k-stages

__device__ __forceinline__ v8bf ld8(const bf16* p) { return *(const v8bf*)p; }

__device__ __forceinline__ v16bf cat16(v8bf a, v8bf b) {
    return __builtin_shufflevector(a, b, 0,1,2,3,4,5,6,7,8,9,10,11,12,13,14,15);
}

__device__ __forceinline__ void split_bf16(float f, bf16& h, bf16& l) {
    h = (bf16)f;                     // RNE: top ~8 mantissa bits
    l = (bf16)(f - (float)h);        // residual: next ~8 mantissa bits
}

// 1-D contiguous f32 tile copy, global -> LDS, via the Tensor Data Mover.
// D# group0/group1 built per CDNA5 ISA 8.3/8.4; groups 2/3 zero (<=2D tensor).
// This toolchain's builtin is the 6-arg form: (g0, g1, g2, g3, g_extra, cpol).
__device__ __forceinline__ void tdm_load_f32_1d(const float* gptr, float* lptr, uint32_t n)
{
    const uint64_t ga = (uint64_t)(uintptr_t)gptr;        // 57-bit global byte address
    const uint32_t la = (uint32_t)(uintptr_t)lptr;        // low 32 bits = LDS byte offset

    v4u g0;
    g0[0] = 1u;                                           // count=1, user descriptor
    g0[1] = la;                                           // lds_addr [63:32]
    g0[2] = (uint32_t)ga;                                 // global_addr [95:64]
    g0[3] = (uint32_t)((ga >> 32) & 0x01FFFFFFu)          // global_addr [120:96]
          | (2u << 30);                                   // type=2 ("image") [127:126]

    v8i g1;
    g1[0] = (int)(2u << 16);                              // workgroup_mask=0, data_size=2 (4B)
    g1[1] = (int)((n & 0xFFFFu) << 16);                   // tensor_dim0 lo16 @ [63:48]
    g1[2] = (int)((n >> 16) & 0xFFFFu);                   // tensor_dim0 hi16 @ [79:64]
    g1[3] = (int)((n & 0xFFFFu) << 16);                   // tile_dim0 @ [127:112]
    g1[4] = 0; g1[5] = 0; g1[6] = 0; g1[7] = 0;           // tile_dim1/2=0 (unused), strides=0

    const v4i gz4 = {0, 0, 0, 0};
    const v8i gz8 = {0, 0, 0, 0, 0, 0, 0, 0};
    __builtin_amdgcn_tensor_load_to_lds(g0, g1, gz4, gz4, gz8, 0);
}

// ---------------------------------------------------------------------------
// Kernel 1: xa[M,16] = x[M,K] @ A[16,K]^T   (f32, tiny: ~1 GFLOP)
// lane == row -> A addresses are wave-uniform (scalar loads); 16 f32 acc/lane.
// ---------------------------------------------------------------------------
__global__ __launch_bounds__(128)
void lora_xa_kernel(const float* __restrict__ X, const float* __restrict__ A,
                    float* __restrict__ XA)
{
    const int lane = threadIdx.x & 31;
    const int wv   = threadIdx.x >> 5;
    const int row  = blockIdx.x * 128 + wv * 32 + lane;

    float acc[RANK];
#pragma unroll
    for (int r = 0; r < RANK; ++r) acc[r] = 0.0f;

    const float* xrow = X + (size_t)row * K_DIM;
    for (int k = 0; k < K_DIM; k += 4) {
        float4 xv = *(const float4*)(xrow + k);
#pragma unroll
        for (int r = 0; r < RANK; ++r) {
            float4 av = *(const float4*)(A + (size_t)r * K_DIM + k);
            acc[r] += xv.x * av.x + xv.y * av.y + xv.z * av.z + xv.w * av.w;
        }
    }
#pragma unroll
    for (int r = 0; r < RANK; r += 4) {
        float4 o;
        o.x = acc[r]; o.y = acc[r + 1]; o.z = acc[r + 2]; o.w = acc[r + 3];
        *(float4*)(XA + (size_t)row * RANK + r) = o;
    }
}

// ---------------------------------------------------------------------------
// Kernel 2: main fused GEMM. Block = 128x128 out tile, 256 threads (8 wave32).
// Wave grid 2(m) x 4(n); wave tile 64x32 = 4x2 WMMA accumulators.
// ---------------------------------------------------------------------------
__global__ __launch_bounds__(256)
void lora_gemm_wmma(const float* __restrict__ X, const float* __restrict__ W,
                    const float* __restrict__ bias, const float* __restrict__ Blora,
                    const float* __restrict__ XA, float* __restrict__ out)
{
    __shared__ bf16 sXh[2][BM * LDA];
    __shared__ bf16 sXl[2][BM * LDA];
    __shared__ bf16 sWh[2][BN * LDA];
    __shared__ bf16 sWl[2][BN * LDA];
    __shared__ float sXa[BM * RANK];   // epilogue: xa tile   (TDM target)
    __shared__ float sBl[BN * RANK];   // epilogue: LoRA B tile (TDM target)

    const int tid  = threadIdx.x;
    const int lane = tid & 31;
    const int wv   = tid >> 5;
    const int mw   = wv & 1;          // 2 wave-rows of 64
    const int nw   = wv >> 1;         // 4 wave-cols of 32
    const int m0   = blockIdx.y * BM;
    const int n0   = blockIdx.x * BN;
    const int ln   = lane & 15;

    v8f acc[4][2];
#pragma unroll
    for (int i = 0; i < 4; ++i) { acc[i][0] = {}; acc[i][1] = {}; }

    float4 rx[4], rw[4];   // global->LDS staging (16 f32 of x, 16 of W per thread)

    auto load_global = [&](int kt) {
        const int kbase = kt * BK;
#pragma unroll
        for (int i = 0; i < 4; ++i) {
            int idx = tid + i * 256;
            int r   = idx >> 3;          // 8 float4 per 32-wide row
            int c   = (idx & 7) << 2;
            rx[i] = *(const float4*)(X + (size_t)(m0 + r) * K_DIM + kbase + c);
            rw[i] = *(const float4*)(W + (size_t)(n0 + r) * K_DIM + kbase + c);
        }
    };

    auto store_lds = [&](int buf) {
#pragma unroll
        for (int i = 0; i < 4; ++i) {
            int idx = tid + i * 256;
            int r   = idx >> 3;
            int c   = (idx & 7) << 2;
            float fx[4] = {rx[i].x, rx[i].y, rx[i].z, rx[i].w};
            float fw[4] = {rw[i].x, rw[i].y, rw[i].z, rw[i].w};
            v4bf vxh, vxl, vwh, vwl;
#pragma unroll
            for (int e = 0; e < 4; ++e) {
                bf16 h, l;
                split_bf16(fx[e], h, l); vxh[e] = h; vxl[e] = l;
                split_bf16(fw[e], h, l); vwh[e] = h; vwl[e] = l;
            }
            const int off = r * LDA + c;             // 8B-aligned
            *(v4bf*)&sXh[buf][off] = vxh;
            *(v4bf*)&sXl[buf][off] = vxl;
            *(v4bf*)&sWh[buf][off] = vwh;
            *(v4bf*)&sWl[buf][off] = vwl;
        }
    };

    auto compute = [&](int buf) {
        const int ca = (lane < 16) ? 0 : 8;    // A frag k-chunks: {ca, ca+16}
        const int cb = (lane < 16) ? 0 : 8;    // B frag k-chunks: {cb, cb+16}
        v16bf bh[2], bl[2];
#pragma unroll
        for (int ni = 0; ni < 2; ++ni) {
            const int n = nw * 32 + ni * 16 + ln;
            const bf16* ph = &sWh[buf][n * LDA + cb];
            const bf16* pl = &sWl[buf][n * LDA + cb];
            bh[ni] = cat16(ld8(ph), ld8(ph + 16));
            bl[ni] = cat16(ld8(pl), ld8(pl + 16));
        }
#pragma unroll
        for (int mi = 0; mi < 4; ++mi) {
            const int m = mw * 64 + mi * 16 + ln;
            const bf16* ph = &sXh[buf][m * LDA + ca];
            const bf16* pl = &sXl[buf][m * LDA + ca];
            v16bf ah = cat16(ld8(ph), ld8(ph + 16));
            v16bf al = cat16(ld8(pl), ld8(pl + 16));
#pragma unroll
            for (int ni = 0; ni < 2; ++ni) {
                v8f c = acc[mi][ni];
                c = __builtin_amdgcn_wmma_f32_16x16x32_bf16(false, ah, false, bh[ni], (short)0, c, false, false);
                c = __builtin_amdgcn_wmma_f32_16x16x32_bf16(false, ah, false, bl[ni], (short)0, c, false, false);
                c = __builtin_amdgcn_wmma_f32_16x16x32_bf16(false, al, false, bh[ni], (short)0, c, false, false);
                acc[mi][ni] = c;
            }
        }
    };

    // ---- k-loop: double-buffered LDS, global loads overlapped with compute ----
    load_global(0);
    store_lds(0);
    __syncthreads();

    for (int kt = 0; kt < NK; ++kt) {
        const int cur = kt & 1;
        const bool has_next = (kt + 1) < NK;
        if (has_next) load_global(kt + 1);
        if (kt + 4 < NK) {                      // deep prefetch -> global_prefetch_b8
            const int kb = (kt + 4) * BK;
            const int r  = tid >> 1;
            const int c  = (tid & 1) * 16;
            __builtin_prefetch(X + (size_t)(m0 + r) * K_DIM + kb + c, 0, 1);
            __builtin_prefetch(W + (size_t)(n0 + r) * K_DIM + kb + c, 0, 1);
        }
        compute(cur);
        if (has_next) store_lds(cur ^ 1);
        __syncthreads();

        // Kick off the epilogue TDM copies early: issue once, by wave 0. The
        // targets never alias the stage buffers; completion awaited below.
        if (kt == 0 && wv == 0) {
            tdm_load_f32_1d(XA    + (size_t)m0 * RANK, sXa, BM * RANK);   // 8 KB
            tdm_load_f32_1d(Blora + (size_t)n0 * RANK, sBl, BN * RANK);   // 8 KB
        }
    }

    // ---- epilogue: wait for TDM tiles, fuse bias + rank-16 LoRA delta ----
    if (wv == 0) {
        __builtin_amdgcn_s_wait_tensorcnt(0);   // wave 0 owns TENSORcnt
    }
    __syncthreads();                            // publishes LDS tiles to all waves

    const int mh = (lane >> 4) << 3;            // 0 or 8 (C/D layout M offset)
    float bcol[2][RANK];
    float bs[2];
#pragma unroll
    for (int ni = 0; ni < 2; ++ni) {
        const int n = nw * 32 + ni * 16 + ln;
#pragma unroll
        for (int r = 0; r < RANK; ++r) bcol[ni][r] = sBl[n * RANK + r];
        bs[ni] = bias[n0 + n];
    }

#pragma unroll
    for (int mi = 0; mi < 4; ++mi) {
#pragma unroll
        for (int j = 0; j < 8; ++j) {
            const int mloc = mw * 64 + mi * 16 + mh + j;
            const float* xr = &sXa[mloc * RANK];
            float xv[RANK];
#pragma unroll
            for (int r = 0; r < RANK; ++r) xv[r] = xr[r];
#pragma unroll
            for (int ni = 0; ni < 2; ++ni) {
                float d = 0.0f;
#pragma unroll
                for (int r = 0; r < RANK; ++r) d += xv[r] * bcol[ni][r];
                const int n = nw * 32 + ni * 16 + ln;
                out[(size_t)(m0 + mloc) * N_DIM + n0 + n] =
                    acc[mi][ni][j] + bs[ni] + LORA_SCALE * d;
            }
        }
    }
}

// ---------------------------------------------------------------------------
extern "C" void kernel_launch(void* const* d_in, const int* in_sizes, int n_in,
                              void* d_out, int out_size, void* d_ws, size_t ws_size,
                              hipStream_t stream)
{
    const float* x    = (const float*)d_in[0];   // [4,2048,4096] f32
    const float* W    = (const float*)d_in[1];   // [4096,4096]   f32
    const float* bias = (const float*)d_in[2];   // [4096]        f32
    const float* A    = (const float*)d_in[3];   // [16,4096]     f32
    const float* B    = (const float*)d_in[4];   // [4096,16]     f32
    float* out = (float*)d_out;                  // [8192,4096]   f32
    float* xa  = (float*)d_ws;                   // [8192,16]     f32 (512 KB scratch)

    lora_xa_kernel<<<M_DIM / 128, 128, 0, stream>>>(x, A, xa);

    dim3 grid(N_DIM / BN, M_DIM / BM);           // 32 x 64 = 2048 blocks
    lora_gemm_wmma<<<grid, 256, 0, stream>>>(x, W, bias, B, xa, out);
}
